// TexualEmbeddingLayer_41609643163657
// MI455X (gfx1250) — matrix operand order repaired and checked by reading, not compiled, not run on hardware
//
#include <hip/hip_runtime.h>

// ---------------- problem constants ----------------
#define BS       64
#define SEQ      1024
#define IN_DIM   512
#define EMB      1024
#define HID      512
#define K_TOP    306
#define KPAD     320                 // pad K to multiple of 64 for GEMM tiling
#define MTOT     (BS * KPAD)         // 20480 rows (multiple of 128)
#define MASK_VALUE (-10000.0f)
#define BN_EPS   1e-5f

typedef __attribute__((ext_vector_type(16))) _Float16 v16h;
typedef __attribute__((ext_vector_type(8)))  float    v8f;

union V16U { v16h h; uint4 u[2]; };
static_assert(sizeof(V16U) == 32, "v16h must be 32 bytes");

// CDNA5 async global->LDS staging path (gfx1250), with safe fallback
#if defined(__gfx1250__) && __has_builtin(__builtin_amdgcn_global_load_async_to_lds_b128) && __has_builtin(__builtin_amdgcn_s_wait_asynccnt)
#define USE_ASYNC_LDS 1
#else
#define USE_ASYNC_LDS 0
#endif

#if USE_ASYNC_LDS
typedef int v4i_vs __attribute__((vector_size(16)));   // matches builtin param pointee
#define ASYNC_CP16(g, l)                                                          \
    __builtin_amdgcn_global_load_async_to_lds_b128(                               \
        (__attribute__((address_space(1))) v4i_vs*)(uintptr_t)(g),                \
        (__attribute__((address_space(3))) v4i_vs*)(unsigned)(uintptr_t)(l), 0, 0)
#endif

// ---------------- kernel 0: zero BN stats ----------------
__global__ void zero_kernel(float* p, int n) {
    int i = blockIdx.x * blockDim.x + threadIdx.x;
    if (i < n) p[i] = 0.0f;
}

// ---------------- kernel 1: token lengths / eos position ----------------
__global__ void lens_kernel(const int* __restrict__ text, int* lens, int* eos) {
    int b = blockIdx.x;
    __shared__ int red[256];
    int cnt = 0;
    for (int i = threadIdx.x; i < SEQ; i += 256)
        cnt += (text[b * SEQ + i] != 0) ? 1 : 0;
    red[threadIdx.x] = cnt;
    __syncthreads();
    for (int s = 128; s > 0; s >>= 1) {
        if (threadIdx.x < s) red[threadIdx.x] += red[threadIdx.x + s];
        __syncthreads();
    }
    if (threadIdx.x == 0) {
        int L = red[0] > 0 ? red[0] : 1;
        lens[b] = L;
        int e = L - 1; if (e < 0) e = 0; if (e > SEQ - 1) e = SEQ - 1;
        eos[b] = e;
    }
}

// ---------------- kernel 2: one-pass column softmax + eos-row select ------------
// softmax over dim=1 (rows); each thread owns one column, streams 1024 rows.
__global__ void softmax_sel_kernel(const float* __restrict__ atten,
                                   const int* __restrict__ text,
                                   const int* __restrict__ eos_p,
                                   float* __restrict__ sel) {
    int b = blockIdx.y;
    int c = blockIdx.x * blockDim.x + threadIdx.x;
    int eos = eos_p[b];
    bool padcol = (text[b * SEQ + c] == 0);
    bool maskedcol = (c == 0) || (c == eos) || padcol;
    const float* A = atten + (size_t)b * SEQ * SEQ + c;
    float m = -INFINITY, s = 0.0f, v = MASK_VALUE;
    for (int r = 0; r < SEQ; ++r) {
        float x = A[(size_t)r * SEQ];
        __builtin_prefetch(&A[(size_t)(r + 8) * SEQ], 0, 1);
        if (!isfinite(x)) x = 0.0f;              // nan_to_num(nan/±inf -> 0)
        if (maskedcol) x = MASK_VALUE;
        if (r == eos) v = x;
        float nm = fmaxf(m, x);
        s = s * __expf(m - nm) + __expf(x - nm);
        m = nm;
    }
    float r = __expf(v - m) / s;
    sel[b * SEQ + c] = padcol ? 0.0f : r;
}

// ---------------- kernel 3: per-batch top-K via bitonic sort (stable ties) ------
__global__ void topk_kernel(const float* __restrict__ sel, int* __restrict__ topk) {
    int b = blockIdx.x;
    __shared__ float key[SEQ];
    __shared__ int   idx[SEQ];
    for (int i = threadIdx.x; i < SEQ; i += blockDim.x) {
        key[i] = sel[b * SEQ + i];
        idx[i] = i;
    }
    __syncthreads();
    for (int k = 2; k <= SEQ; k <<= 1) {
        for (int j = k >> 1; j > 0; j >>= 1) {
            for (int t = threadIdx.x; t < SEQ; t += blockDim.x) {
                int ixj = t ^ j;
                if (ixj > t) {
                    float a = key[t], c2 = key[ixj];
                    int ia = idx[t], ic = idx[ixj];
                    // "greater" = larger value, or equal value with smaller index
                    bool agtc = (a > c2) || (a == c2 && ia < ic);
                    bool descending = ((t & k) == 0);
                    bool sw = descending ? !agtc : agtc;
                    if (sw) { key[t] = c2; key[ixj] = a; idx[t] = ic; idx[ixj] = ia; }
                }
            }
            __syncthreads();
        }
    }
    for (int i = threadIdx.x; i < KPAD; i += blockDim.x)
        topk[b * KPAD + i] = (i < K_TOP) ? idx[i] : 0;
}

// ---------------- kernel 4: gather selected features, L2-normalize, f16 --------
__global__ void gather_norm_kernel(const float* __restrict__ feats,
                                   const int* __restrict__ topk,
                                   _Float16* __restrict__ X) {
    int b = blockIdx.y, kk = blockIdx.x;
    __shared__ float red[256];
    size_t orow = ((size_t)b * KPAD + kk) * IN_DIM;
    if (kk >= K_TOP) {   // zero padding rows so GEMM tiles are clean
        for (int d = threadIdx.x; d < IN_DIM; d += 256) X[orow + d] = (_Float16)0.0f;
        return;
    }
    int src = topk[b * KPAD + kk];
    const float* f = feats + ((size_t)b * SEQ + src) * IN_DIM;
    float ss = 0.0f;
    float vals[2];
#pragma unroll
    for (int i = 0; i < 2; ++i) {
        int d = threadIdx.x + i * 256;
        vals[i] = f[d];
        ss += vals[i] * vals[i];
    }
    red[threadIdx.x] = ss;
    __syncthreads();
    for (int s = 128; s > 0; s >>= 1) {
        if (threadIdx.x < s) red[threadIdx.x] += red[threadIdx.x + s];
        __syncthreads();
    }
    float inv = 1.0f / fmaxf(sqrtf(red[0]), 1e-6f);
#pragma unroll
    for (int i = 0; i < 2; ++i) {
        int d = threadIdx.x + i * 256;
        X[orow + d] = (_Float16)(vals[i] * inv);
    }
}

// ---------------- kernel 5: transpose+convert weights (K x N f32 -> N x K f16) --
__global__ void wcvt_kernel(const float* __restrict__ W, _Float16* __restrict__ Wt,
                            int Kd, int N) {
    int i = blockIdx.x * blockDim.x + threadIdx.x;
    if (i >= Kd * N) return;
    int n = i / Kd, k = i % Kd;
    Wt[i] = (_Float16)W[(size_t)k * N + n];
}

// ---------------- kernel 6: WMMA GEMM: D = A(MxKd,f16) * Bt^T + bias (+Cadd) ----
// Block: 128 threads = 4 waves; tile 128(M) x 64(N); wave tile 32x64; K-step 32.
#define LDT 40   // LDS row stride in halves (80 B, 16 B aligned, de-skewed)

__global__ void __launch_bounds__(128)
gemm_wmma_kernel(const _Float16* __restrict__ A, const _Float16* __restrict__ Bt,
                 const float* __restrict__ bias, const float* __restrict__ Cadd,
                 float* __restrict__ D, int N, int Kd,
                 float* __restrict__ bnSum, float* __restrict__ bnSqSum) {
    __shared__ _Float16 sA[128 * LDT];
    __shared__ _Float16 sB[64 * LDT];
    int m0 = blockIdx.y * 128;
    int n0 = blockIdx.x * 64;
    int tid = threadIdx.x;
    int wave = tid >> 5, lane = tid & 31;

    v8f acc[2][4];
#pragma unroll
    for (int a = 0; a < 2; ++a)
#pragma unroll
        for (int t = 0; t < 4; ++t)
#pragma unroll
            for (int v = 0; v < 8; ++v) acc[a][t][v] = 0.0f;

    // fragment lane mapping (ISA 7.12.2)
    int arow = lane & 15;
    int kb   = (lane >> 4) * 8;          // A K-base: 0 or 8
    int nl   = lane & 15;
    int koff = (lane >> 4) * 8;          // B K-offset: 0 or 8

    // staging mapping: thread t stages A row t (4 x 16B) and half of B row t/2 (2 x 16B)
    const _Float16* gA = A + (size_t)(m0 + tid) * Kd;
    _Float16*       lA = sA + tid * LDT;
    const _Float16* gB = Bt + (size_t)(n0 + (tid >> 1)) * Kd + (tid & 1) * 16;
    _Float16*       lB = sB + (tid >> 1) * LDT + (tid & 1) * 16;

    for (int k0 = 0; k0 < Kd; k0 += 32) {
#if USE_ASYNC_LDS
        // CDNA5 async copy straight into LDS (ASYNCcnt-tracked, no VGPR round-trip)
        ASYNC_CP16(gA + k0,      lA);
        ASYNC_CP16(gA + k0 + 8,  lA + 8);
        ASYNC_CP16(gA + k0 + 16, lA + 16);
        ASYNC_CP16(gA + k0 + 24, lA + 24);
        ASYNC_CP16(gB + k0,      lB);
        ASYNC_CP16(gB + k0 + 8,  lB + 8);
        __builtin_amdgcn_s_wait_asynccnt(0);
#else
        {
            const uint4* sa = (const uint4*)(gA + k0);
            uint4* da = (uint4*)lA;
            da[0] = sa[0]; da[1] = sa[1]; da[2] = sa[2]; da[3] = sa[3];
            const uint4* sb = (const uint4*)(gB + k0);
            uint4* db = (uint4*)lB;
            db[0] = sb[0]; db[1] = sb[1];
        }
#endif
        __syncthreads();

        // A fragments for this wave's two 16-row sub-tiles
        V16U av0, av1;
        const _Float16* pa0 = sA + (wave * 32 + arow) * LDT;
        const _Float16* pa1 = pa0 + 16 * LDT;
        av0.u[0] = *(const uint4*)(pa0 + kb);
        av0.u[1] = *(const uint4*)(pa0 + kb + 16);
        av1.u[0] = *(const uint4*)(pa1 + kb);
        av1.u[1] = *(const uint4*)(pa1 + kb + 16);

#pragma unroll
        for (int t = 0; t < 4; ++t) {
            V16U bv;
            const _Float16* bp = sB + (t * 16 + nl) * LDT + koff;
            bv.u[0] = *(const uint4*)bp;
            bv.u[1] = *(const uint4*)(bp + 16);
            acc[0][t] = __builtin_amdgcn_wmma_f32_16x16x32_f16(
                false, av0.h, false, bv.h, (short)0, acc[0][t], false, false);
            acc[1][t] = __builtin_amdgcn_wmma_f32_16x16x32_f16(
                false, av1.h, false, bv.h, (short)0, acc[1][t], false, false);
        }
        __syncthreads();
    }

    // epilogue: lane l writes rows (sub-tile base) + (l>>4)*8 + v, col n0+t*16+(l&15)
    int rhalf = (lane >> 4) * 8;
#pragma unroll
    for (int a = 0; a < 2; ++a) {
        int mw = m0 + wave * 32 + a * 16;
#pragma unroll
        for (int t = 0; t < 4; ++t) {
            int ncol = n0 + t * 16 + nl;
            float bco = bias ? bias[ncol] : 0.0f;
            float lsum = 0.0f, lsq = 0.0f;
#pragma unroll
            for (int v = 0; v < 8; ++v) {
                int row = mw + rhalf + v;
                float val = acc[a][t][v] + bco;
                if (Cadd) val += Cadd[(size_t)row * N + ncol];
                D[(size_t)row * N + ncol] = val;
                if (bnSum && ((row % KPAD) < K_TOP)) { lsum += val; lsq += val * val; }
            }
            if (bnSum) {
                atomicAdd(&bnSum[ncol], lsum);
                atomicAdd(&bnSqSum[ncol], lsq);
            }
        }
    }
}

// ---------------- kernel 7: BatchNorm (batch stats) + ReLU + f16 convert --------
__global__ void bn_relu_kernel(const float* __restrict__ h,
                               const float* __restrict__ bnSum,
                               const float* __restrict__ bnSq,
                               const float* __restrict__ g,
                               const float* __restrict__ be,
                               _Float16* __restrict__ h16) {
    size_t i = (size_t)blockIdx.x * blockDim.x + threadIdx.x;
    size_t total = (size_t)MTOT * HID;
    if (i >= total) return;
    int n = (int)(i % HID);
    const float cnt = (float)(BS * K_TOP);
    float mu  = bnSum[n] / cnt;
    float var = bnSq[n] / cnt - mu * mu;     // biased variance
    float x = (h[i] - mu) * rsqrtf(var + BN_EPS) * g[n] + be[n];
    h16[i] = (_Float16)fmaxf(x, 0.0f);
}

// ---------------- kernel 8: variable-length top-1 (max) pooling -----------------
__global__ void pool_kernel(const float* __restrict__ fused,
                            const int* __restrict__ lens,
                            float* __restrict__ out) {
    int b = blockIdx.y;
    int e = blockIdx.x * blockDim.x + threadIdx.x;
    int pl = lens[b] - 2;
    if (pl < 1) pl = 1;
    if (pl > K_TOP) pl = K_TOP;
    const float* F = fused + (size_t)b * KPAD * EMB + e;
    float m = -INFINITY;
    for (int k = 0; k < pl; ++k) m = fmaxf(m, F[(size_t)k * EMB]);
    out[b * EMB + e] = m;
}

// ---------------- host launcher ----------------
extern "C" void kernel_launch(void* const* d_in, const int* in_sizes, int n_in,
                              void* d_out, int out_size, void* d_ws, size_t ws_size,
                              hipStream_t stream) {
    (void)in_sizes; (void)n_in; (void)out_size; (void)ws_size;
    const float* features = (const float*)d_in[0];   // (64,1024,512)
    const float* atten    = (const float*)d_in[1];   // (64,1024,1024)
    const int*   text     = (const int*)d_in[2];     // (64,1024) int32
    const float* lin_w    = (const float*)d_in[3];   // (512,1024)
    const float* lin_b    = (const float*)d_in[4];   // (1024,)
    const float* w1       = (const float*)d_in[5];   // (512,512)
    const float* b1       = (const float*)d_in[6];   // (512,)
    const float* g1       = (const float*)d_in[7];   // (512,)
    const float* be1      = (const float*)d_in[8];   // (512,)
    const float* w2       = (const float*)d_in[9];   // (512,1024)
    const float* b2       = (const float*)d_in[10];  // (1024,)
    float* out = (float*)d_out;                      // (64,1024)

    // workspace carve-out (256 B aligned)
    char* w = (char*)d_ws;
    size_t off = 0;
    auto alloc = [&](size_t bytes) -> void* {
        void* p = w + off;
        off += (bytes + 255) & ~(size_t)255;
        return p;
    };
    int*      lens   = (int*)alloc(BS * sizeof(int));
    int*      eos    = (int*)alloc(BS * sizeof(int));
    float*    sel    = (float*)alloc((size_t)BS * SEQ * sizeof(float));
    int*      topk   = (int*)alloc((size_t)BS * KPAD * sizeof(int));
    _Float16* X16    = (_Float16*)alloc((size_t)MTOT * IN_DIM * sizeof(_Float16));
    _Float16* linT   = (_Float16*)alloc((size_t)EMB * IN_DIM * sizeof(_Float16));
    _Float16* w1T    = (_Float16*)alloc((size_t)HID * IN_DIM * sizeof(_Float16));
    _Float16* w2T    = (_Float16*)alloc((size_t)EMB * HID * sizeof(_Float16));
    float*    hbuf   = (float*)alloc((size_t)MTOT * HID * sizeof(float));
    _Float16* h16    = (_Float16*)alloc((size_t)MTOT * HID * sizeof(_Float16));
    float*    fused  = (float*)alloc((size_t)MTOT * EMB * sizeof(float));
    float*    bnstat = (float*)alloc(2 * HID * sizeof(float));
    float* bnSum = bnstat;
    float* bnSq  = bnstat + HID;

    // 0. zero BN stats (workspace is poisoned, not zeroed)
    zero_kernel<<<dim3((2 * HID + 255) / 256), 256, 0, stream>>>(bnstat, 2 * HID);
    // 1. token lengths / eos
    lens_kernel<<<dim3(BS), 256, 0, stream>>>(text, lens, eos);
    // 2. column softmax + eos-row selection (streams 256 MB once)
    softmax_sel_kernel<<<dim3(SEQ / 256, BS), 256, 0, stream>>>(atten, text, eos, sel);
    // 3. per-batch top-K
    topk_kernel<<<dim3(BS), 256, 0, stream>>>(sel, topk);
    // 4. gather + L2 normalize + f16
    gather_norm_kernel<<<dim3(KPAD, BS), 256, 0, stream>>>(features, topk, X16);
    // 5. weight transpose/convert
    wcvt_kernel<<<dim3((IN_DIM * EMB + 255) / 256), 256, 0, stream>>>(lin_w, linT, IN_DIM, EMB);
    wcvt_kernel<<<dim3((IN_DIM * HID + 255) / 256), 256, 0, stream>>>(w1, w1T, IN_DIM, HID);
    wcvt_kernel<<<dim3((HID * EMB + 255) / 256), 256, 0, stream>>>(w2, w2T, HID, EMB);
    // 6a. cap_emb = X @ lin_w + lin_b   (written directly into `fused`)
    gemm_wmma_kernel<<<dim3(EMB / 64, MTOT / 128), 128, 0, stream>>>(
        X16, linT, lin_b, nullptr, fused, EMB, IN_DIM, nullptr, nullptr);
    // 6b. h = X @ w1 + b1  (+ BN partial sums over valid rows)
    gemm_wmma_kernel<<<dim3(HID / 64, MTOT / 128), 128, 0, stream>>>(
        X16, w1T, b1, nullptr, hbuf, HID, IN_DIM, bnSum, bnSq);
    // 7. BN(batch stats) + ReLU -> f16
    {
        size_t total = (size_t)MTOT * HID;
        bn_relu_kernel<<<dim3((unsigned)((total + 255) / 256)), 256, 0, stream>>>(
            hbuf, bnSum, bnSq, g1, be1, h16);
    }
    // 6c. fused = cap_emb + (h16 @ w2 + b2)
    gemm_wmma_kernel<<<dim3(EMB / 64, MTOT / 128), 128, 0, stream>>>(
        h16, w2T, b2, fused, fused, EMB, HID, nullptr, nullptr);
    // 8. variable-length max pool -> out
    pool_kernel<<<dim3(EMB / 256, BS), 256, 0, stream>>>(fused, lens, out);
}
